// OrderInvariantKernelLinearAttention_32770600468500
// MI455X (gfx1250) — compile-verified, matrix operand order checked by reading, not run on hardware
//
#include <hip/hip_runtime.h>
#include <hip/hip_bf16.h>

typedef __attribute__((ext_vector_type(16))) __bf16 v16bf;
typedef __attribute__((ext_vector_type(8)))  float  v8f;

#define S_LEN   4096
#define D_DIM   128
#define BLK     512
#define NB      (S_LEN / BLK)          // 8 blocks per (b,h)
#define H_NUM   32
#define CHUNK   32                     // WMMA K per step (bf16)
#define NCHUNK  (BLK / CHUNK)          // 16
#define KT_STRIDE 40                   // padded LDS row stride (halfs) = 80B, 16B-aligned
#define S_STRIDE  136                  // padded S' row stride (halfs) = 272B, 16B-aligned
#define MT_N    9                      // 8 value column-tiles + 1 ones/den tile

__device__ __forceinline__ unsigned short f2bf(float f) {
  unsigned int u = __builtin_bit_cast(unsigned int, f);
  u += 0x7FFFu + ((u >> 16) & 1u);     // round-to-nearest-even
  return (unsigned short)(u >> 16);
}

// phi(x) = elu(x*s+b)+1  ==  y>0 ? y+1 : exp(y)
__device__ __forceinline__ float phi_f(float x, float s, float b) {
  float y = __builtin_fmaf(x, s, b);
  return y > 0.f ? y + 1.f : __expf(y);
}

// Load one bf16 16x32 fragment: two aligned 16B chunks per lane.
// Lane half h holds K = {8h..8h+7} in elements 0..7, K = {16+8h..23+8h} in 8..15.
__device__ __forceinline__ v16bf frag_lds(const unsigned short* row, int c0) {
  union { v16bf v; uint4 q[2]; } f;
  f.q[0] = *(const uint4*)(row + c0);
  f.q[1] = *(const uint4*)(row + c0 + 16);
  return f.v;
}

__device__ __forceinline__ v8f vzero8() {
  v8f z;
  #pragma unroll
  for (int i = 0; i < 8; ++i) z[i] = 0.f;
  return z;
}

__global__ __launch_bounds__(256)
void linattn_block_kernel(const float* __restrict__ Q, const float* __restrict__ K,
                          const float* __restrict__ V, const float* __restrict__ Sc,
                          const float* __restrict__ Bi, float* __restrict__ Out) {
  __shared__ __align__(16) unsigned short lds_kT[D_DIM * KT_STRIDE];          // phi(K)^T chunk
  __shared__ __align__(16) unsigned short lds_vT[(MT_N * 16) * KT_STRIDE];    // [V|1|0]^T chunk
  __shared__ __align__(16) unsigned short lds_S [(MT_N * 16) * S_STRIDE];     // S'^T (bf16)
  __shared__ __align__(16) float lds_scale[D_DIM];
  __shared__ __align__(16) float lds_bias [D_DIM];

  const int t    = threadIdx.x;        // 0..255 (8 waves of 32)
  const int lane = t & 31;
  const int w    = t >> 5;             // wave id 0..7
  const int nn   = lane & 15;
  const int hh   = lane >> 4;          // lane half

  const int head = blockIdx.x / NB;    // b*H + h
  const int hidx = head % H_NUM;
  const int sblk = (blockIdx.x % NB) * BLK;

  const size_t base = ((size_t)head * S_LEN + sblk) * D_DIM;
  const float* Qp = Q + base;
  const float* Kp = K + base;
  const float* Vp = V + base;
  float*       Op = Out + base;

  if (t < D_DIM) {
    lds_scale[t] = Sc[hidx * D_DIM + t];
    lds_bias[t]  = Bi[hidx * D_DIM + t];
  }
  // Constant rows of [V|1|0]^T: row 128 = ones (Z/den column), rows 129..143 = 0.
  for (int i = t; i < 16 * KT_STRIDE; i += 256) {
    int r = i / KT_STRIDE;
    lds_vT[(D_DIM + r) * KT_STRIDE + (i % KT_STRIDE)] =
        (r == 0) ? (unsigned short)0x3F80u : (unsigned short)0u;
  }
  __syncthreads();

  const int   d_t  = t & 127;          // fixed feature index for staging
  const int   shalf = t >> 7;          // which 16 sequence rows this thread stages
  const float sc_t = lds_scale[d_t];
  const float bi_t = lds_bias[d_t];

  // ---------- Phase A: [S' | Z] = phi(K)^T @ [V | 1 | 0] ----------
  v8f accA[MT_N];
  #pragma unroll
  for (int m = 0; m < MT_N; ++m) accA[m] = vzero8();

  for (int c = 0; c < NCHUNK; ++c) {
    const int s0 = c * CHUNK;
    // Stage transposed bf16 chunk; 8 consecutive s per packed ds_store_b128.
    #pragma unroll
    for (int i = 0; i < 2; ++i) {
      const int sbase = shalf * 16 + 8 * i;            // 0,8 / 16,24
      union { uint4 q; unsigned short u[8]; } pk, pv;
      #pragma unroll
      for (int j = 0; j < 8; ++j) {
        const size_t g = (size_t)(s0 + sbase + j) * D_DIM + d_t;
        pk.u[j] = f2bf(phi_f(Kp[g], sc_t, bi_t));
        pv.u[j] = f2bf(Vp[g]);
      }
      *(uint4*)&lds_kT[d_t * KT_STRIDE + sbase] = pk.q;
      *(uint4*)&lds_vT[d_t * KT_STRIDE + sbase] = pv.q;
      if (c + 1 < NCHUNK) {                            // hint next chunk
        __builtin_prefetch(&Kp[(size_t)(s0 + CHUNK + sbase) * D_DIM + d_t], 0, 0);
        __builtin_prefetch(&Vp[(size_t)(s0 + CHUNK + sbase) * D_DIM + d_t], 0, 0);
      }
    }
    __syncthreads();

    // A fragment: 16 rows d = 16*w + nn of phi(K)^T, K = 32 sequence positions
    const v16bf afr = frag_lds(&lds_kT[(16 * w + nn) * KT_STRIDE], 8 * hh);
    #pragma unroll
    for (int mt = 0; mt < MT_N; ++mt) {
      const v16bf bfr = frag_lds(&lds_vT[(16 * mt + nn) * KT_STRIDE], 8 * hh);
      accA[mt] = __builtin_amdgcn_wmma_f32_16x16x32_bf16(
          false, afr, false, bfr, (short)0, accA[mt], false, false);
    }
    __syncthreads();
  }

  // Stash S'^T (bf16) so phase-B B-fragments read contiguous rows.
  // C/D layout: lane holds column n = nn; VGPR r holds row r + 8*hh.
  #pragma unroll
  for (int mt = 0; mt < MT_N; ++mt) {
    #pragma unroll
    for (int r = 0; r < 8; ++r) {
      const int d = 16 * w + r + 8 * hh;
      const int m = 16 * mt + nn;
      lds_S[m * S_STRIDE + d] = f2bf(accA[mt][r]);
    }
  }
  __syncthreads();

  // ---------- Phase B: [num | den] = phi(Q) @ [S' | Z] ----------
  const float4* sc4 = (const float4*)lds_scale;
  const float4* bi4 = (const float4*)lds_bias;

  for (int it = 0; it < 4; ++it) {
    const int st   = w + 8 * it;                       // s-tile 0..31
    const int srow = 16 * st + nn;                     // q row for this lane
    const float4* qv4 = (const float4*)(Qp + (size_t)srow * D_DIM);

    // A fragments: phi(Q) row, 4 chunks of K=32, vectorized b128 loads.
    v16bf afr[4];
    #pragma unroll
    for (int kc = 0; kc < 4; ++kc) {
      const int d0 = 32 * kc + 8 * hh;                 // multiple of 8
      const int p  = d0 >> 2;
      union { float4 v4[2]; float f[8]; } qa, qb, sa, sb, ba, bb;
      qa.v4[0] = qv4[p];     qa.v4[1] = qv4[p + 1];
      qb.v4[0] = qv4[p + 4]; qb.v4[1] = qv4[p + 5];
      sa.v4[0] = sc4[p];     sa.v4[1] = sc4[p + 1];
      sb.v4[0] = sc4[p + 4]; sb.v4[1] = sc4[p + 5];
      ba.v4[0] = bi4[p];     ba.v4[1] = bi4[p + 1];
      bb.v4[0] = bi4[p + 4]; bb.v4[1] = bi4[p + 5];
      union { v16bf v; unsigned short u[16]; } f;
      #pragma unroll
      for (int j = 0; j < 8; ++j) {
        f.u[j]     = f2bf(phi_f(qa.f[j], sa.f[j], ba.f[j]));
        f.u[j + 8] = f2bf(phi_f(qb.f[j], sb.f[j], bb.f[j]));
      }
      afr[kc] = f.v;
    }

    v8f acc[MT_N];
    #pragma unroll
    for (int m = 0; m < MT_N; ++m) acc[m] = vzero8();

    #pragma unroll
    for (int mt = 0; mt < MT_N; ++mt) {
      #pragma unroll
      for (int kc = 0; kc < 4; ++kc) {
        const v16bf bfr = frag_lds(&lds_S[(16 * mt + nn) * S_STRIDE], 32 * kc + 8 * hh);
        acc[mt] = __builtin_amdgcn_wmma_f32_16x16x32_bf16(
            false, afr[kc], false, bfr, (short)0, acc[mt], false, false);
      }
    }

    // out = num / (den + eps); den = column n=0 of tile 8, lives in lane 0 / 16.
    #pragma unroll
    for (int r = 0; r < 8; ++r) {
      const float den = __shfl(acc[8][r], lane & 16, 32) + 1e-6f;
      const float inv = 1.0f / den;
      const int s_out = 16 * st + r + 8 * hh;
      float* orow = Op + (size_t)s_out * D_DIM;
      #pragma unroll
      for (int mt = 0; mt < 8; ++mt) {
        orow[16 * mt + nn] = acc[mt][r] * inv;
      }
    }
  }
}

extern "C" void kernel_launch(void* const* d_in, const int* in_sizes, int n_in,
                              void* d_out, int out_size, void* d_ws, size_t ws_size,
                              hipStream_t stream) {
  const float* q  = (const float*)d_in[0];
  const float* k  = (const float*)d_in[1];
  const float* v  = (const float*)d_in[2];
  const float* sc = (const float*)d_in[3];
  const float* bi = (const float*)d_in[4];
  float* out = (float*)d_out;

  const int heads  = in_sizes[0] / (S_LEN * D_DIM);   // B*H = 64
  const int blocks = heads * NB;                      // 512 workgroups
  hipLaunchKernelGGL(linattn_block_kernel, dim3(blocks), dim3(256), 0, stream,
                     q, k, v, sc, bi, out);
}
